// UNetSphericalHealpixResidualLongConnections_67869073211457
// MI455X (gfx1250) — compile-verified, matrix operand order checked by reading, not run on hardware
//
#include <hip/hip_runtime.h>
#include <hip/hip_bf16.h>

// ---------------------------------------------------------------------------
// UNet spherical healpix — CDNA5 (gfx1250) implementation.
// Matmuls: bf16 V_WMMA_F32_16X16X32_BF16 (f32 accumulate), one wave = 32x32
// register-blocked tile (2x2 WMMA sub-tiles, A/B fragment reuse).
// Chebyshev K=3 packed as a single [M, 3*Kp] bf16 operand -> one GEMM per conv.
// ---------------------------------------------------------------------------

typedef __attribute__((ext_vector_type(16))) __bf16 v16bf;
typedef __attribute__((ext_vector_type(8)))  __bf16 v8bf;
typedef __attribute__((ext_vector_type(8)))  float  v8f;

#define BN_   16
#define V0N   3072
#define V1N   768
#define V2N   192
#define M0N   (BN_*V0N)   /* 49152 */
#define M1N   (BN_*V1N)   /* 12288 */
#define M2N   (BN_*V2N)   /* 3072  */
#define EPSB  1e-5f

static inline int cdiv_l(long a, int b){ return (int)((a + (long)b - 1) / b); }
static inline int kpad(int f){ return (f + 31) & ~31; }

// ----------------------------- utility kernels -----------------------------

__global__ void k_zero_f32(float* p, long n){
  long i = (long)blockIdx.x*blockDim.x + threadIdx.x;
  if (i < n) p[i] = 0.f;
}
__global__ void k_zero_bf16(__bf16* p, long n){
  long i = (long)blockIdx.x*blockDim.x + threadIdx.x;
  if (i < n) p[i] = (__bf16)0.f;
}

// f32 [M,F] -> bf16 slice [M,Kp] (row stride `stride`), zero-padding cols F..Kp-1
__global__ void k_cvt_pad(const float* __restrict__ src, __bf16* __restrict__ dst,
                          long M, int F, int Kp, int stride){
  long i = (long)blockIdx.x*blockDim.x + threadIdx.x;
  long tot = M * (long)Kp;
  if (i >= tot) return;
  int  kk = (int)(i % Kp);
  long r  = i / Kp;
  float v = (kk < F) ? src[r*(long)F + kk] : 0.f;
  dst[r*(long)stride + kk] = (__bf16)v;
}

// sparse Laplacian apply: val = scale*(diag[v]*y + sum_e w[v,e]*y[nbr]) - (dosub? xm2 : 0)
// writes f32 (for recurrence) and bf16 into a Chebyshev slice.
__global__ void k_lmul(const float* __restrict__ y, const float* __restrict__ w,
                       const int* __restrict__ nbr, const float* __restrict__ diag,
                       const float* __restrict__ xm2,
                       float* __restrict__ outf, __bf16* __restrict__ outb,
                       int Bv, int V, int F, int stride, float scale, int dosub){
  long i = (long)blockIdx.x*blockDim.x + threadIdx.x;
  long tot = (long)Bv*V*F;
  if (i >= tot) return;
  int  f = (int)(i % F);
  long t = i / F;
  int  v = (int)(t % V);
  int  b = (int)(t / V);
  const float* yb = y + (long)b*V*F;
  float acc = diag[v] * yb[(long)v*F + f];
  #pragma unroll
  for (int e = 0; e < 8; ++e){
    int u = nbr[v*8 + e];
    acc += w[v*8 + e] * yb[(long)u*F + f];
  }
  float val = scale * acc;
  if (dosub) val -= xm2[i];
  outf[i] = val;
  outb[((long)b*V + v)*(long)stride + f] = (__bf16)val;
}

// cheb weight W[3][fin][fout] -> Wt[fout][3*Kp] bf16, column-major over K, pad zeros
__global__ void k_wt_cheb(const float* __restrict__ W, __bf16* __restrict__ Wt,
                          int fin, int fout, int Kp){
  long i = (long)blockIdx.x*blockDim.x + threadIdx.x;
  long tot = (long)fout * 3 * Kp;
  if (i >= tot) return;
  int  kk = (int)(i % Kp);
  long t  = i / Kp;
  int  kc = (int)(t % 3);
  int  n  = (int)(t / 3);
  float v = (kk < fin) ? W[((long)kc*fin + kk)*fout + n] : 0.f;
  Wt[i] = (__bf16)v;
}

// res weight W[fout][fin] -> Wt[fout][Kp] bf16 (already K-major), pad zeros
__global__ void k_wt_res(const float* __restrict__ W, __bf16* __restrict__ Wt,
                         int fin, int fout, int Kp){
  long i = (long)blockIdx.x*blockDim.x + threadIdx.x;
  long tot = (long)fout * Kp;
  if (i >= tot) return;
  int kk = (int)(i % Kp);
  int n  = (int)(i / Kp);
  float v = (kk < fin) ? W[(long)n*fin + kk] : 0.f;
  Wt[i] = (__bf16)v;
}

// --------------------------- WMMA bf16 GEMM --------------------------------
// C[M,N] = A[M,Kdim](bf16, row stride lda) * Bt[N,Kdim]^T (bf16) + bias
// One wave -> 32x32 register-blocked tile = 2x2 WMMA 16x16 sub-tiles.
// Lane mapping per ISA 7.12.2 (16-bit A/B fragments, wave32):
//   lanes 0-15: K = 0..7 / 16..23 ; lanes 16-31: K = 8..15 / 24..31 per k-step.
// Out-of-range B columns are clamped to row 0 of Bt (results discarded at
// store time) -> no EXEC manipulation in the inner loop.

__device__ __forceinline__ v16bf ld_frag(const __bf16* p){
  v8bf lo = *(const v8bf*)(p);
  v8bf hi = *(const v8bf*)(p + 16);
  return __builtin_shufflevector(lo, hi, 0,1,2,3,4,5,6,7,8,9,10,11,12,13,14,15);
}
__device__ __forceinline__ v8f wmma_bf16(v16bf a, v16bf b, v8f c){
  return __builtin_amdgcn_wmma_f32_16x16x32_bf16(false, a, false, b, (short)0, c, false, false);
}

__global__ void __launch_bounds__(128)
k_gemm_wmma(const __bf16* __restrict__ A, int lda,
            const __bf16* __restrict__ Bt, int Kdim,
            const float* __restrict__ bias,
            float* __restrict__ C, int M, int N){
  const int wave = threadIdx.x >> 5;
  const int lane = threadIdx.x & 31;
  const int tm   = blockIdx.x << 5;                   // 32 rows per block
  const int tn   = (blockIdx.y << 7) + (wave << 5);   // 32 cols per wave
  if (tn >= N) return;                                // uniform per-wave exit

  const int mn    = lane & 15;
  const int half  = lane >> 4;
  const int ko    = half << 3;
  const int col0  = tn + mn;
  const int col1  = tn + 16 + mn;
  const bool ok0  = (col0 < N);
  const bool ok1  = (col1 < N);

  v8f acc00, acc01, acc10, acc11;
  {
    float i0 = (ok0 && bias) ? bias[col0] : 0.f;
    float i1 = (ok1 && bias) ? bias[col1] : 0.f;
    #pragma unroll
    for (int r = 0; r < 8; ++r){ acc00[r] = i0; acc01[r] = i1; acc10[r] = i0; acc11[r] = i1; }
  }

  const __bf16* a0 = A  + (long)(tm + mn)      * lda + ko;
  const __bf16* a1 = A  + (long)(tm + 16 + mn) * lda + ko;
  const __bf16* b0 = Bt + (long)(ok0 ? col0 : 0) * Kdim + ko;
  const __bf16* b1 = Bt + (long)(ok1 ? col1 : 0) * Kdim + ko;

  for (int kb = 0; kb < Kdim; kb += 32){
    // pull next K-chunks toward the WGP while this one computes
    __builtin_prefetch((const void*)(a0 + kb + 128), 0, 0);
    __builtin_prefetch((const void*)(a1 + kb + 128), 0, 0);
    __builtin_prefetch((const void*)(b0 + kb + 128), 0, 0);
    __builtin_prefetch((const void*)(b1 + kb + 128), 0, 0);
    v16bf av0 = ld_frag(a0 + kb);
    v16bf av1 = ld_frag(a1 + kb);
    v16bf bv0 = ld_frag(b0 + kb);
    v16bf bv1 = ld_frag(b1 + kb);
    acc00 = wmma_bf16(av0, bv0, acc00);
    acc01 = wmma_bf16(av0, bv1, acc01);
    acc10 = wmma_bf16(av1, bv0, acc10);
    acc11 = wmma_bf16(av1, bv1, acc11);
  }

  // C/D layout: VGPR r <-> row (8*half + r), column lane&15 within sub-tile
  if (ok0){
    #pragma unroll
    for (int r = 0; r < 8; ++r){
      C[(long)(tm +      (half << 3) + r) * N + col0] = acc00[r];
      C[(long)(tm + 16 + (half << 3) + r) * N + col0] = acc10[r];
    }
  }
  if (ok1){
    #pragma unroll
    for (int r = 0; r < 8; ++r){
      C[(long)(tm +      (half << 3) + r) * N + col1] = acc01[r];
      C[(long)(tm + 16 + (half << 3) + r) * N + col1] = acc11[r];
    }
  }
}

// ------------------------------- batchnorm ---------------------------------

__global__ void k_bn_stats(const float* __restrict__ h, int M, int N,
                           float* __restrict__ mean, float* __restrict__ invstd){
  const int c = blockIdx.x;
  __shared__ float s1[256];
  __shared__ float s2[256];
  float a = 0.f, b = 0.f;
  for (int r = threadIdx.x; r < M; r += 256){
    float v = h[(long)r*N + c];
    a += v; b += v*v;
  }
  s1[threadIdx.x] = a; s2[threadIdx.x] = b;
  __syncthreads();
  for (int s = 128; s > 0; s >>= 1){
    if ((int)threadIdx.x < s){
      s1[threadIdx.x] += s1[threadIdx.x + s];
      s2[threadIdx.x] += s2[threadIdx.x + s];
    }
    __syncthreads();
  }
  if (threadIdx.x == 0){
    float m   = s1[0] / (float)M;
    float var = s2[0] / (float)M - m*m;
    mean[c]   = m;
    invstd[c] = rsqrtf(var + EPSB);
  }
}

__global__ void k_bn_apply_relu(float* __restrict__ h, long n, int N,
                                const float* __restrict__ mean, const float* __restrict__ invstd,
                                const float* __restrict__ gamma, const float* __restrict__ beta){
  long i = (long)blockIdx.x*blockDim.x + threadIdx.x;
  if (i >= n) return;
  int c = (int)(i % N);
  float v = (h[i] - mean[c]) * invstd[c] * gamma[c] + beta[c];
  h[i] = v > 0.f ? v : 0.f;
}

__global__ void k_add(const float* __restrict__ a, const float* __restrict__ b,
                      float* __restrict__ o, long n){
  long i = (long)blockIdx.x*blockDim.x + threadIdx.x;
  if (i < n) o[i] = a[i] + b[i];
}

// ------------------------------ pool / unpool ------------------------------

__global__ void k_pool(const float* __restrict__ x, float* __restrict__ vals,
                       int* __restrict__ idx, int Bv, int Vc, int F){
  long i = (long)blockIdx.x*blockDim.x + threadIdx.x;
  long tot = (long)Bv*Vc*F;
  if (i >= tot) return;
  int  f  = (int)(i % F);
  long t  = i / F;
  int  vc = (int)(t % Vc);
  int  b  = (int)(t / Vc);
  const int Vf = Vc * 4;
  long base = ((long)b*Vf + vc*4)*F + f;
  float best = x[base]; int am = 0;
  #pragma unroll
  for (int p = 1; p < 4; ++p){
    float v = x[base + (long)p*F];
    if (v > best){ best = v; am = p; }
  }
  vals[i] = best;
  idx[i]  = vc*4 + am;
}

__global__ void k_unpool(const float* __restrict__ x, const int* __restrict__ idx,
                         float* __restrict__ out, int Bv, int Vc, int F, int Vout){
  long i = (long)blockIdx.x*blockDim.x + threadIdx.x;
  long tot = (long)Bv*Vc*F;
  if (i >= tot) return;
  int  f  = (int)(i % F);
  long t  = i / F;
  int  b  = (int)(t / Vc);
  out[((long)b*Vout + idx[i])*F + f] = x[i];
}

__global__ void k_copycols(const float* __restrict__ src, float* __restrict__ dst,
                           long M, int Fs, int Fd, int off){
  long i = (long)blockIdx.x*blockDim.x + threadIdx.x;
  long tot = M * (long)Fs;
  if (i >= tot) return;
  int  c = (int)(i % Fs);
  long r = i / Fs;
  dst[r*(long)Fd + off + c] = src[i];
}

// ---------------------------- host-side helpers ----------------------------

static void run_cheb(hipStream_t s, const float* xin, int Bv, int V, int F,
                     const float* lw, const int* lnbr, const float* ldiag,
                     __bf16* Xb, float* t1, float* t2,
                     const __bf16* Wt, const float* bias, float* C, int N){
  const long M = (long)Bv * V;
  const int  Kp = kpad(F);
  const int  stride = 3 * Kp;
  if (Kp != F){
    long n = M * (long)stride;
    k_zero_bf16<<<cdiv_l(n,256),256,0,s>>>(Xb, n);
  }
  k_cvt_pad<<<cdiv_l(M*(long)Kp,256),256,0,s>>>(xin, Xb, M, F, Kp, stride);
  k_lmul<<<cdiv_l(M*(long)F,256),256,0,s>>>(xin, lw, lnbr, ldiag, nullptr,
                                            t1, Xb + Kp,   Bv, V, F, stride, 1.f, 0);
  k_lmul<<<cdiv_l(M*(long)F,256),256,0,s>>>(t1, lw, lnbr, ldiag, xin,
                                            t2, Xb + 2*Kp, Bv, V, F, stride, 2.f, 1);
  dim3 g((unsigned)(M/32), (unsigned)((N + 127) / 128));
  k_gemm_wmma<<<g,128,0,s>>>(Xb, stride, Wt, 3*Kp, bias, C, (int)M, N);
}

static void run_bn_relu(hipStream_t s, float* h, long M, int N,
                        const float* gamma, const float* beta,
                        float* mean, float* invstd){
  k_bn_stats<<<N,256,0,s>>>(h, (int)M, N, mean, invstd);
  k_bn_apply_relu<<<cdiv_l(M*(long)N,256),256,0,s>>>(h, M*(long)N, N, mean, invstd, gamma, beta);
}

// -------------------------------- entry -----------------------------------

extern "C" void kernel_launch(void* const* d_in, const int* in_sizes, int n_in,
                              void* d_out, int out_size, void* d_ws, size_t ws_size,
                              hipStream_t stream){
  (void)in_sizes; (void)n_in; (void)out_size; (void)ws_size;

  // ---- input leaves (setup_inputs insertion order) ----
  const float* x    = (const float*)d_in[0];
  const float *W11=(const float*)d_in[1],  *b11=(const float*)d_in[2],  *g11=(const float*)d_in[3],  *be11=(const float*)d_in[4];
  const float *W13=(const float*)d_in[5],  *b13=(const float*)d_in[6],  *g13=(const float*)d_in[7],  *be13=(const float*)d_in[8];
  const float *Wr1=(const float*)d_in[9],  *br1=(const float*)d_in[10];
  const float *W21=(const float*)d_in[11], *b21=(const float*)d_in[12], *g21=(const float*)d_in[13], *be21=(const float*)d_in[14];
  const float *W23=(const float*)d_in[15], *b23=(const float*)d_in[16], *g23=(const float*)d_in[17], *be23=(const float*)d_in[18];
  const float *Wr2=(const float*)d_in[19], *br2=(const float*)d_in[20];
  const float *W31=(const float*)d_in[21], *b31=(const float*)d_in[22], *g31=(const float*)d_in[23], *be31=(const float*)d_in[24];
  const float *W33=(const float*)d_in[25], *b33=(const float*)d_in[26], *g33=(const float*)d_in[27], *be33=(const float*)d_in[28];
  const float *Wr3=(const float*)d_in[29], *br3=(const float*)d_in[30];
  const float *Wu21=(const float*)d_in[31],*bu21=(const float*)d_in[32],*gu21=(const float*)d_in[33],*beu21=(const float*)d_in[34];
  const float *Wu22=(const float*)d_in[35],*bu22=(const float*)d_in[36],*gu22=(const float*)d_in[37],*beu22=(const float*)d_in[38];
  const float *Wu11=(const float*)d_in[39],*bu11=(const float*)d_in[40],*gu11=(const float*)d_in[41],*beu11=(const float*)d_in[42];
  const float *Wu12=(const float*)d_in[43],*bu12=(const float*)d_in[44],*gu12=(const float*)d_in[45],*beu12=(const float*)d_in[46];
  const float *Wu13=(const float*)d_in[47],*bu13=(const float*)d_in[48];
  const int   *nbr0=(const int*)d_in[49];  const float *lw0=(const float*)d_in[50], *ld0=(const float*)d_in[51];
  const int   *nbr1=(const int*)d_in[52];  const float *lw1=(const float*)d_in[53], *ld1=(const float*)d_in[54];
  const int   *nbr2=(const int*)d_in[55];  const float *lw2=(const float*)d_in[56], *ld2=(const float*)d_in[57];

  float* out = (float*)d_out;                       // [B, V0, 4]

  // ---- deterministic bump allocator over d_ws ----
  char*  ws  = (char*)d_ws;
  size_t off = 0;
  auto alloc = [&](size_t bytes)->void*{
    off = (off + 255) & ~(size_t)255;
    void* p = ws + off;
    off += bytes;
    return p;
  };
  auto af = [&](long n){ return (float*) alloc((size_t)n*4); };
  auto ai = [&](long n){ return (int*)   alloc((size_t)n*4); };
  auto ab = [&](long n){ return (__bf16*)alloc((size_t)n*2); };

  float* e11  = af((long)M0N*64);
  float* e1   = af((long)M0N*128);
  float* e2   = af((long)M1N*256);
  float* e2i  = af((long)M1N*128);
  float* e3i  = af((long)M2N*256);
  float* h    = af((long)M0N*128);   // GEMM / block scratch (max live size)
  float* res  = af((long)M0N*128);   // residual scratch; reused as `up` in decoder
  float* tmpA = af((long)M0N*256);
  float* tmpB = af((long)M0N*256);
  float* cat  = af((long)M0N*256);
  float* mean = af(512);
  float* invs = af(512);
  int*   idx1 = ai((long)M1N*128);
  int*   idx2 = ai((long)M2N*256);
  __bf16* Xb  = ab((long)M0N*768);   // Chebyshev operand [M, 3*Kp], max level-0 Kp=256

  __bf16 *wt11 = ab(64*3*32),   *wt13 = ab(128*3*64),  *wtr1 = ab(128*32);
  __bf16 *wt21 = ab(192*3*128), *wt23 = ab(256*3*192), *wtr2 = ab(256*128);
  __bf16 *wt31 = ab(512*3*256), *wt33 = ab(256*3*512), *wtr3 = ab(256*256);
  __bf16 *wtu21= ab(256*3*512), *wtu22= ab(128*3*256);
  __bf16 *wtu11= ab(128*3*256), *wtu12= ab(64*3*128),  *wtu13= ab(4*3*128);

  // ---- weight prep (f32 -> bf16, K-major, padded) ----
  auto prep_cheb = [&](const float* W, __bf16* Wt, int fin, int fout){
    int Kp = kpad(fin); long n = (long)fout*3*Kp;
    k_wt_cheb<<<cdiv_l(n,256),256,0,stream>>>(W, Wt, fin, fout, Kp);
  };
  auto prep_res = [&](const float* W, __bf16* Wt, int fin, int fout){
    int Kp = kpad(fin); long n = (long)fout*Kp;
    k_wt_res<<<cdiv_l(n,256),256,0,stream>>>(W, Wt, fin, fout, Kp);
  };
  prep_cheb(W11,  wt11,  16, 64);   prep_cheb(W13,  wt13,  64, 128);  prep_res(Wr1, wtr1, 16, 128);
  prep_cheb(W21,  wt21, 128, 192);  prep_cheb(W23,  wt23, 192, 256);  prep_res(Wr2, wtr2, 128, 256);
  prep_cheb(W31,  wt31, 256, 512);  prep_cheb(W33,  wt33, 512, 256);  prep_res(Wr3, wtr3, 256, 256);
  prep_cheb(Wu21, wtu21,512, 256);  prep_cheb(Wu22, wtu22,256, 128);
  prep_cheb(Wu11, wtu11,256, 128);  prep_cheb(Wu12, wtu12,128, 64);   prep_cheb(Wu13, wtu13,128, 4);

  auto res_gemm = [&](__bf16* XbA, int Kp, const __bf16* Wt, const float* bias,
                      float* C, long M, int N){
    dim3 g((unsigned)(M/32), (unsigned)((N + 127) / 128));
    k_gemm_wmma<<<g,128,0,stream>>>(XbA, 3*Kp, Wt, Kp, bias, C, (int)M, N);
  };

  // ======================= encoder level 0 (V=3072) ========================
  run_cheb(stream, x, BN_, V0N, 16, lw0, nbr0, ld0, Xb, tmpA, tmpB, wt11, b11, e11, 64);
  res_gemm(Xb, 32, wtr1, br1, res, M0N, 128);                 // res(x) while Xb slice0 == x
  run_bn_relu(stream, e11, M0N, 64, g11, be11, mean, invs);   // e11 persists (skip connection)
  run_cheb(stream, e11, BN_, V0N, 64, lw0, nbr0, ld0, Xb, tmpA, tmpB, wt13, b13, h, 128);
  run_bn_relu(stream, h, M0N, 128, g13, be13, mean, invs);
  k_add<<<cdiv_l((long)M0N*128,256),256,0,stream>>>(h, res, e1, (long)M0N*128);

  // ======================= encoder level 1 (V=768) =========================
  k_pool<<<cdiv_l((long)M1N*128,256),256,0,stream>>>(e1, e2i, idx1, BN_, V1N, 128);
  run_cheb(stream, e2i, BN_, V1N, 128, lw1, nbr1, ld1, Xb, tmpA, tmpB, wt21, b21, h, 192);
  res_gemm(Xb, 128, wtr2, br2, res, M1N, 256);                // res(e2i)
  run_bn_relu(stream, h, M1N, 192, g21, be21, mean, invs);
  run_cheb(stream, h, BN_, V1N, 192, lw1, nbr1, ld1, Xb, tmpA, tmpB, wt23, b23, h, 256);
  run_bn_relu(stream, h, M1N, 256, g23, be23, mean, invs);
  k_add<<<cdiv_l((long)M1N*256,256),256,0,stream>>>(h, res, e2, (long)M1N*256);

  // ======================= encoder level 2 (V=192) =========================
  k_pool<<<cdiv_l((long)M2N*256,256),256,0,stream>>>(e2, e3i, idx2, BN_, V2N, 256);
  run_cheb(stream, e3i, BN_, V2N, 256, lw2, nbr2, ld2, Xb, tmpA, tmpB, wt31, b31, h, 512);
  res_gemm(Xb, 256, wtr3, br3, res, M2N, 256);                // res(e3i)
  run_bn_relu(stream, h, M2N, 512, g31, be31, mean, invs);
  run_cheb(stream, h, BN_, V2N, 512, lw2, nbr2, ld2, Xb, tmpA, tmpB, wt33, b33, h, 256);
  run_bn_relu(stream, h, M2N, 256, g33, be33, mean, invs);
  k_add<<<cdiv_l((long)M2N*256,256),256,0,stream>>>(h, res, h, (long)M2N*256);  // e3 in h

  // ======================= decoder level 1 (V=768) =========================
  float* up = res;  // residual scratch is dead from here on
  k_zero_f32<<<cdiv_l((long)M1N*256,256),256,0,stream>>>(up, (long)M1N*256);
  k_unpool<<<cdiv_l((long)M2N*256,256),256,0,stream>>>(h, idx2, up, BN_, V2N, 256, V1N);
  k_copycols<<<cdiv_l((long)M1N*256,256),256,0,stream>>>(up, cat, M1N, 256, 512, 0);
  k_copycols<<<cdiv_l((long)M1N*256,256),256,0,stream>>>(e2, cat, M1N, 256, 512, 256);
  run_cheb(stream, cat, BN_, V1N, 512, lw1, nbr1, ld1, Xb, tmpA, tmpB, wtu21, bu21, h, 256);
  run_bn_relu(stream, h, M1N, 256, gu21, beu21, mean, invs);
  run_cheb(stream, h, BN_, V1N, 256, lw1, nbr1, ld1, Xb, tmpA, tmpB, wtu22, bu22, h, 128);
  run_bn_relu(stream, h, M1N, 128, gu22, beu22, mean, invs);

  // ======================= decoder level 0 (V=3072) ========================
  k_zero_f32<<<cdiv_l((long)M0N*128,256),256,0,stream>>>(up, (long)M0N*128);
  k_unpool<<<cdiv_l((long)M1N*128,256),256,0,stream>>>(h, idx1, up, BN_, V1N, 128, V0N);
  k_copycols<<<cdiv_l((long)M0N*128,256),256,0,stream>>>(up, cat, M0N, 128, 256, 0);
  k_copycols<<<cdiv_l((long)M0N*128,256),256,0,stream>>>(e1, cat, M0N, 128, 256, 128);
  run_cheb(stream, cat, BN_, V0N, 256, lw0, nbr0, ld0, Xb, tmpA, tmpB, wtu11, bu11, h, 128);
  run_bn_relu(stream, h, M0N, 128, gu11, beu11, mean, invs);
  run_cheb(stream, h, BN_, V0N, 128, lw0, nbr0, ld0, Xb, tmpA, tmpB, wtu12, bu12, h, 64);
  run_bn_relu(stream, h, M0N, 64, gu12, beu12, mean, invs);

  // final: cheb([u, e11]) -> out [M0, 4]  (no BN / relu)
  k_copycols<<<cdiv_l((long)M0N*64,256),256,0,stream>>>(h,   cat, M0N, 64, 128, 0);
  k_copycols<<<cdiv_l((long)M0N*64,256),256,0,stream>>>(e11, cat, M0N, 64, 128, 64);
  run_cheb(stream, cat, BN_, V0N, 128, lw0, nbr0, ld0, Xb, tmpA, tmpB, wtu13, bu13, out, 4);
}
